// MultiHeadAttention_28870770164279
// MI455X (gfx1250) — matrix operand order repaired and checked by reading, not compiled, not run on hardware
//
#include <hip/hip_runtime.h>
#include <hip/hip_bf16.h>
#include <cstdint>

// ---------------- problem constants ----------------
constexpr int Bc  = 2;
constexpr int Sc  = 2048;
constexpr int Dc  = 2048;
constexpr int Hc  = 16;
constexpr int DHc = 128;          // head dim
constexpr float SCALE = 0.08838834764831845f; // DH^-0.5

// ---------------- vector types ----------------
typedef __attribute__((ext_vector_type(16))) __bf16          v16bf;
typedef __attribute__((ext_vector_type(8)))  float           v8f;
typedef __attribute__((ext_vector_type(8)))  unsigned short  u16x8;
typedef __attribute__((ext_vector_type(16))) unsigned short  u16x16;

static __device__ inline unsigned short f2bf(float f) {
    unsigned u = __builtin_bit_cast(unsigned, f);
    unsigned r = u + 0x7FFFu + ((u >> 16) & 1u);   // round-to-nearest-even
    return (unsigned short)(r >> 16);
}

static __device__ inline v16bf mk_frag(u16x8 lo, u16x8 hi) {
    u16x16 t;
#pragma unroll
    for (int i = 0; i < 8; ++i) { t[i] = lo[i]; t[i + 8] = hi[i]; }
    return __builtin_bit_cast(v16bf, t);
}

static __device__ inline v8f wmma_bf16(v16bf a, v16bf b, v8f c) {
    return __builtin_amdgcn_wmma_f32_16x16x32_bf16(false, a, false, b,
                                                   (short)0, c, false, false);
}

// gfx1250 async data mover: per-lane 16B global -> LDS copy, tracked by ASYNCcnt.
static __device__ inline void async_g2l_b128(unsigned lds_off, const void* gaddr) {
    unsigned long long ga = (unsigned long long)(uintptr_t)gaddr;
    asm volatile("global_load_async_to_lds_b128 %0, %1, off"
                 :: "v"(lds_off), "v"(ga) : "memory");
}
static __device__ inline void wait_async0() {
    asm volatile("s_wait_asynccnt 0x0" ::: "memory");
}

// ---------------- fp32 -> bf16 conversion ----------------
__global__ __launch_bounds__(256)
void f32_to_bf16_kernel(const float* __restrict__ src,
                        unsigned short* __restrict__ dst, long long n4) {
    long long i = (long long)blockIdx.x * 256 + threadIdx.x;
    if (i < n4) {
        float4 f = ((const float4*)src)[i];
        ushort4 o;
        o.x = f2bf(f.x); o.y = f2bf(f.y); o.z = f2bf(f.z); o.w = f2bf(f.w);
        ((ushort4*)dst)[i] = o;
    }
}

// ---------------- tiled WMMA GEMM: C(f32) = A(bf16 MxK) * B(bf16 KxN) + bias ----
// block = 128 threads (4 waves), tile M=64 N=64, k-step 32.
__global__ __launch_bounds__(128)
void gemm_bf16_kernel(const unsigned short* __restrict__ A,
                      const unsigned short* __restrict__ Bm,
                      const float* __restrict__ bias,
                      float* __restrict__ C,
                      int M, int N, int K) {
    __shared__ unsigned short As[64][40];   // [m][k], padded
    __shared__ unsigned short Bt[64][40];   // [n][k], transposed, padded

    const int tid  = threadIdx.x;
    const int wave = tid >> 5;
    const int lane = tid & 31;
    const int lr   = lane & 15;
    const int half = (lane < 16) ? 0 : 1;
    const int m0 = blockIdx.y * 64;
    const int n0 = blockIdx.x * 64;

    v8f acc[4];
#pragma unroll
    for (int i = 0; i < 4; ++i)
#pragma unroll
        for (int j = 0; j < 8; ++j) acc[i][j] = 0.f;

    for (int k0 = 0; k0 < K; k0 += 32) {
        // stage A tile 64x32: async global->LDS (no transpose needed)
#pragma unroll
        for (int rep = 0; rep < 2; ++rep) {
            int v = tid + rep * 128;
            int row = v >> 2, cg = (v & 3) * 8;
            async_g2l_b128((unsigned)(uintptr_t)&As[row][cg],
                           A + (size_t)(m0 + row) * K + k0 + cg);
        }
        // stage B tile 32x64 transposed (regular path: transpose in the store)
#pragma unroll
        for (int rep = 0; rep < 2; ++rep) {
            int v = tid + rep * 128;
            int kk = v >> 3, ng = (v & 7) * 8;
            u16x8 d = *(const u16x8*)(Bm + (size_t)(k0 + kk) * N + n0 + ng);
#pragma unroll
            for (int i = 0; i < 8; ++i) Bt[ng + i][kk] = d[i];
        }
        wait_async0();
        __syncthreads();

        // A fragment: lane row = wave*16+lr; K = kb..kb+7 and kb+16..kb+23
        const int arow = wave * 16 + lr;
        const int kb   = half ? 8 : 0;
        u16x8 alo = *(const u16x8*)(&As[arow][kb]);
        u16x8 ahi = *(const u16x8*)(&As[arow][kb + 16]);
        v16bf af = mk_frag(alo, ahi);

        const int kg = half ? 8 : 0;
#pragma unroll
        for (int nf = 0; nf < 4; ++nf) {
            int bn = nf * 16 + lr;
            u16x8 blo = *(const u16x8*)(&Bt[bn][kg]);
            u16x8 bhi = *(const u16x8*)(&Bt[bn][kg + 16]);
            v16bf bf = mk_frag(blo, bhi);
            acc[nf] = wmma_bf16(af, bf, acc[nf]);
        }
        __syncthreads();
    }

    // epilogue: C layout -> VGPR r holds M = r + half*8, col = lane%16
#pragma unroll
    for (int nf = 0; nf < 4; ++nf) {
        int col = n0 + nf * 16 + lr;
        float bv = bias[col];
#pragma unroll
        for (int r = 0; r < 8; ++r) {
            int row = m0 + wave * 16 + r + half * 8;
            C[(size_t)row * N + col] = acc[nf][r] + bv;
        }
    }
}

// ---------------- RoPE + split into q/k/v bf16 [B,H,S,DH] ----------------
// NOTE: reference broadcasts cos[:, :H] against the HEAD axis -> angle index = h.
__global__ __launch_bounds__(256)
void rope_split_kernel(const float* __restrict__ qkv,
                       const float* __restrict__ fcos,
                       const float* __restrict__ fsin,
                       unsigned short* __restrict__ qb,
                       unsigned short* __restrict__ kb,
                       unsigned short* __restrict__ vb) {
    const int bs = blockIdx.x;            // 0 .. B*S-1
    const int b  = bs / Sc, s = bs % Sc;
    const int f  = threadIdx.x & 63;
    const int h0 = threadIdx.x >> 6;      // 0..3
    const float* row = qkv + (size_t)bs * (3 * Dc);
    for (int h = h0; h < Hc; h += 4) {
        const float c  = fcos[h * 64 + f];   // quirk: position index == head
        const float sn = fsin[h * 64 + f];
        float qre = row[h * DHc + f],            qim = row[h * DHc + 64 + f];
        float kre = row[Dc + h * DHc + f],       kim = row[Dc + h * DHc + 64 + f];
        float v0  = row[2 * Dc + h * DHc + f],   v1  = row[2 * Dc + h * DHc + 64 + f];
        size_t base = ((size_t)(b * Hc + h) * Sc + s) * DHc;
        qb[base + f]      = f2bf(qre * c - qim * sn);
        qb[base + 64 + f] = f2bf(qre * sn + qim * c);
        kb[base + f]      = f2bf(kre * c - kim * sn);
        kb[base + 64 + f] = f2bf(kre * sn + kim * c);
        vb[base + f]      = f2bf(v0);
        vb[base + 64 + f] = f2bf(v1);
    }
}

// ---------------- flash attention ----------------
// block = 128 threads (4 waves); each wave owns 16 q rows; KV tile = 32 keys.
__global__ __launch_bounds__(128)
void attn_kernel(const unsigned short* __restrict__ qb,
                 const unsigned short* __restrict__ kbuf,
                 const unsigned short* __restrict__ vbuf,
                 unsigned short* __restrict__ ob) {
    __shared__ unsigned short Ks[32][136];      // [key][d]
    __shared__ unsigned short Vt[128][40];      // [d][key] transposed
    __shared__ unsigned short Ps[4][16][40];    // per-wave P transpose buffer

    const int tid  = threadIdx.x;
    const int wave = tid >> 5;
    const int lane = tid & 31;
    const int lr   = lane & 15;
    const int half = (lane < 16) ? 0 : 1;

    const int nqt = Sc / 64;
    const int bh  = blockIdx.x / nqt;
    const int qt  = blockIdx.x % nqt;
    const int h   = bh % Hc;
    const int b   = bh / Hc;
    const int q0  = qt * 64;
    const int qrow0 = q0 + wave * 16;
    const size_t headbase = (size_t)bh * Sc * DHc;

    // load Q A-fragments (4 d-steps of 32)
    v16bf qf[4];
    {
        const unsigned short* qrow = qb + headbase + (size_t)(qrow0 + lr) * DHc;
        const int kb2 = half ? 8 : 0;
#pragma unroll
        for (int ds = 0; ds < 4; ++ds) {
            u16x8 lo = *(const u16x8*)(qrow + ds * 32 + kb2);
            u16x8 hi = *(const u16x8*)(qrow + ds * 32 + kb2 + 16);
            qf[ds] = mk_frag(lo, hi);
        }
    }

    v8f acc[8];
    float mrow[8], lrow[8];
#pragma unroll
    for (int od = 0; od < 8; ++od)
#pragma unroll
        for (int j = 0; j < 8; ++j) acc[od][j] = 0.f;
#pragma unroll
    for (int r = 0; r < 8; ++r) { mrow[r] = -1e30f; lrow[r] = 0.f; }

    const int kg  = half ? 8 : 0;
    const int njt = q0 / 32 + 2;          // causal: only tiles intersecting diag
    for (int j = 0; j < njt; ++j) {
        const int kv0 = j * 32;
        // cooperative load: K via async global->LDS, V transposed (regular path)
        {
            const int key = tid >> 2;
            const int dg  = (tid & 3) * 32;
            const unsigned short* kr = kbuf + headbase + (size_t)(kv0 + key) * DHc + dg;
            const unsigned short* vr = vbuf + headbase + (size_t)(kv0 + key) * DHc + dg;
#pragma unroll
            for (int c = 0; c < 4; ++c) {
                async_g2l_b128((unsigned)(uintptr_t)&Ks[key][dg + c * 8], kr + c * 8);
                u16x8 dv = *(const u16x8*)(vr + c * 8);
#pragma unroll
                for (int i = 0; i < 8; ++i) Vt[dg + c * 8 + i][key] = dv[i];
            }
        }
        wait_async0();
        __syncthreads();

        // scores for two 16-key subtiles
        v8f s0, s1;
#pragma unroll
        for (int jj = 0; jj < 8; ++jj) { s0[jj] = 0.f; s1[jj] = 0.f; }
#pragma unroll
        for (int ds = 0; ds < 4; ++ds) {
            u16x8 lo = *(const u16x8*)(&Ks[lr][ds * 32 + kg]);
            u16x8 hi = *(const u16x8*)(&Ks[lr][ds * 32 + kg + 16]);
            s0 = wmma_bf16(qf[ds], mk_frag(lo, hi), s0);
            lo = *(const u16x8*)(&Ks[16 + lr][ds * 32 + kg]);
            hi = *(const u16x8*)(&Ks[16 + lr][ds * 32 + kg + 16]);
            s1 = wmma_bf16(qf[ds], mk_frag(lo, hi), s1);
        }

        // scale + causal mask + online softmax (row lives in one 16-lane half)
#pragma unroll
        for (int r = 0; r < 8; ++r) {
            const int qabs = qrow0 + r + half * 8;
            float a0 = s0[r] * SCALE; if (kv0 + lr      > qabs) a0 = -1e38f;
            float a1 = s1[r] * SCALE; if (kv0 + 16 + lr > qabs) a1 = -1e38f;
            float t = fmaxf(a0, a1);
#pragma unroll
            for (int m = 1; m < 16; m <<= 1) t = fmaxf(t, __shfl_xor(t, m, 32));
            float newm = fmaxf(mrow[r], t);
            float cf   = __expf(mrow[r] - newm);
            float p0   = __expf(a0 - newm);
            float p1   = __expf(a1 - newm);
            float rs = p0 + p1;
#pragma unroll
            for (int m = 1; m < 16; m <<= 1) rs += __shfl_xor(rs, m, 32);
            lrow[r] = lrow[r] * cf + rs;
            mrow[r] = newm;
#pragma unroll
            for (int od = 0; od < 8; ++od) acc[od][r] *= cf;
            const int M = r + half * 8;
            Ps[wave][M][lr]      = f2bf(p0);
            Ps[wave][M][16 + lr] = f2bf(p1);
        }

        // P (16x32) x V (32x16 per od) -> acc
        {
            const int kb2 = half ? 8 : 0;
            u16x8 plo = *(const u16x8*)(&Ps[wave][lr][kb2]);
            u16x8 phi = *(const u16x8*)(&Ps[wave][lr][kb2 + 16]);
            v16bf pf = mk_frag(plo, phi);
#pragma unroll
            for (int od = 0; od < 8; ++od) {
                u16x8 vlo = *(const u16x8*)(&Vt[od * 16 + lr][kg]);
                u16x8 vhi = *(const u16x8*)(&Vt[od * 16 + lr][kg + 16]);
                acc[od] = wmma_bf16(pf, mk_frag(vlo, vhi), acc[od]);
            }
        }
        __syncthreads();
    }

    // normalize + store bf16 to [B,S,D] (col = h*DH + d)
#pragma unroll
    for (int od = 0; od < 8; ++od) {
#pragma unroll
        for (int r = 0; r < 8; ++r) {
            float val = acc[od][r] / lrow[r];
            int row = qrow0 + r + half * 8;
            int col = h * DHc + od * 16 + lr;
            ob[(size_t)(b * Sc + row) * Dc + col] = f2bf(val);
        }
    }
}

// ---------------- host-side launch ----------------
extern "C" void kernel_launch(void* const* d_in, const int* in_sizes, int n_in,
                              void* d_out, int out_size, void* d_ws, size_t ws_size,
                              hipStream_t stream) {
    const float* x     = (const float*)d_in[0];
    const float* Wqkv  = (const float*)d_in[1];
    const float* bqkv  = (const float*)d_in[2];
    const float* Wout  = (const float*)d_in[3];
    const float* bout  = (const float*)d_in[4];
    const float* fcos  = (const float*)d_in[5];
    const float* fsin  = (const float*)d_in[6];
    // d_in[7] = mask (bool) — causal mask is computed analytically in-kernel.

    const size_t n_x    = (size_t)Bc * Sc * Dc;          // 8.39M
    const size_t n_wqkv = (size_t)Dc * 3 * Dc;           // 12.58M
    const size_t n_wout = (size_t)Dc * Dc;               // 4.19M
    const size_t n_qkv  = (size_t)Bc * Sc * 3 * Dc;      // 25.17M
    const size_t n_head = (size_t)Bc * Hc * Sc * DHc;    // 8.39M

    char* ws = (char*)d_ws;
    size_t off = 0;
    auto alloc = [&](size_t bytes) { char* p = ws + off; off = (off + bytes + 255) & ~(size_t)255; return p; };

    unsigned short* x_bf    = (unsigned short*)alloc(n_x * 2);     // reused as attn output
    unsigned short* wqkv_bf = (unsigned short*)alloc(n_wqkv * 2);
    unsigned short* wout_bf = (unsigned short*)alloc(n_wout * 2);
    float*          qkv_f   = (float*)alloc(n_qkv * 4);
    unsigned short* q_bf    = (unsigned short*)alloc(n_head * 2);
    unsigned short* k_bf    = (unsigned short*)alloc(n_head * 2);
    unsigned short* v_bf    = (unsigned short*)alloc(n_head * 2);
    unsigned short* attn_bf = x_bf;   // alias: x_bf dead after GEMM1

    // 1) conversions
    f32_to_bf16_kernel<<<(int)((n_x / 4 + 255) / 256), 256, 0, stream>>>(x, x_bf, (long long)(n_x / 4));
    f32_to_bf16_kernel<<<(int)((n_wqkv / 4 + 255) / 256), 256, 0, stream>>>(Wqkv, wqkv_bf, (long long)(n_wqkv / 4));
    f32_to_bf16_kernel<<<(int)((n_wout / 4 + 255) / 256), 256, 0, stream>>>(Wout, wout_bf, (long long)(n_wout / 4));

    // 2) qkv = x @ Wqkv + bqkv   (M=4096, N=6144, K=2048)
    {
        dim3 grid(3 * Dc / 64, (Bc * Sc) / 64);
        gemm_bf16_kernel<<<grid, 128, 0, stream>>>(x_bf, wqkv_bf, bqkv, qkv_f,
                                                   Bc * Sc, 3 * Dc, Dc);
    }

    // 3) RoPE + split to head-major bf16
    rope_split_kernel<<<Bc * Sc, 256, 0, stream>>>(qkv_f, fcos, fsin, q_bf, k_bf, v_bf);

    // 4) flash attention
    attn_kernel<<<Bc * Hc * (Sc / 64), 128, 0, stream>>>(q_bf, k_bf, v_bf, attn_bf);

    // 5) out = attn @ Wout + bout  (M=4096, N=2048, K=2048)
    {
        dim3 grid(Dc / 64, (Bc * Sc) / 64);
        gemm_bf16_kernel<<<grid, 128, 0, stream>>>(attn_bf, wout_bf, bout,
                                                   (float*)d_out, Bc * Sc, Dc, Dc);
    }
}
